// HopfieldHCL_82721070121669
// MI455X (gfx1250) — compile-verified, matrix operand order checked by reading, not run on hardware
//
#include <hip/hip_runtime.h>
#include <hip/hip_bf16.h>

// ---------- CDNA5 WMMA types ----------
typedef __attribute__((ext_vector_type(16))) __bf16   v16bf;
typedef __attribute__((ext_vector_type(8)))  float    v8f;
typedef __attribute__((ext_vector_type(8)))  unsigned v8u;

#define CH   64
#define HW   65536        // 256*256
#define IMGW 256
#define NUM_ITERS 5

// packed bf16 pair {lo=sign(x), hi=sign(y)} as +/-1  (sign(0) treated as +1;
// exact zeros have measure zero for these random-normal-derived values)
__device__ __forceinline__ unsigned pk2sgn(float x, float y){
  return 0x3F803F80u
       | ((__float_as_uint(x) >> 16) & 0x8000u)
       |  (__float_as_uint(y)        & 0x80000000u);
}
// b * sign(a)  via sign-bit transfer
__device__ __forceinline__ float sgnmul(float b, float a){
  return __uint_as_float(__float_as_uint(b) ^ (__float_as_uint(a) & 0x80000000u));
}

#define WMMA_BF16(A,B,C) \
  __builtin_amdgcn_wmma_f32_16x16x32_bf16(false,(A),false,(B),(short)0,(C),false,false)

// ======================================================================
// Kernel 1: direct 3x3 SAME conv, 64->64 channels, f32, LDS-tiled.
// ======================================================================
__global__ __launch_bounds__(256) void conv3x3_kernel(
    const float* __restrict__ x,    // [64][256][256]
    const float* __restrict__ w,    // [64][64][3][3]
    const float* __restrict__ bias, // [64]
    float* __restrict__ q)          // [64][256][256]
{
  __shared__ float sW[8*64*9];
  __shared__ float sX[18*18];
  const int tid = threadIdx.x;
  const int tx = tid & 15, ty = tid >> 4;
  const int coutBase = blockIdx.z * 8;

  for (int i = tid; i < 8*64*9; i += 256)
    sW[i] = w[coutBase*64*9 + i];                    // ((co*64+cin)*9 + k)

  float acc[8];
  #pragma unroll
  for (int i = 0; i < 8; ++i) acc[i] = 0.f;

  const int h0 = blockIdx.y * 16, w0 = blockIdx.x * 16;
  for (int cin = 0; cin < 64; ++cin){
    __syncthreads();
    for (int i = tid; i < 18*18; i += 256){
      int lh = i / 18, lw = i % 18;
      int gh = h0 + lh - 1, gw = w0 + lw - 1;
      float v = 0.f;
      if (gh >= 0 && gh < 256 && gw >= 0 && gw < 256)
        v = x[cin*HW + gh*IMGW + gw];
      sX[i] = v;
    }
    __syncthreads();
    #pragma unroll
    for (int co = 0; co < 8; ++co){
      const float* wr = &sW[(co*64 + cin)*9];
      float a = acc[co];
      #pragma unroll
      for (int kh = 0; kh < 3; ++kh)
        #pragma unroll
        for (int kw = 0; kw < 3; ++kw)
          a = fmaf(wr[kh*3 + kw], sX[(ty+kh)*18 + tx + kw], a);
      acc[co] = a;
    }
  }
  const int p = (h0 + ty)*IMGW + (w0 + tx);
  #pragma unroll
  for (int co = 0; co < 8; ++co)
    q[(coutBase + co)*HW + p] = acc[co] + bias[coutBase + co];
}

// ======================================================================
// Kernel 2: Hopfield recurrence via v_wmma_f32_16x16x32_bf16.
//   New[d,p] = sum_e W[c][d,e] * S[e,p];  S = sign(New) fed back.
//   A = W[c] tile (16d x 32e) as exact hi/lo bf16 split (built once/wave).
//   B = state (32e x 16p) bf16 +/-1. The D layout of M-tile pair (a0,a1)
//   IS the B layout of K-chunk 0 for the next iteration (a2,a3 -> chunk 1),
//   so the recurrence stays in registers; sign+repack is pure bit math.
//   TWO pixel tiles are processed per loop iteration (sharing A fragments):
//   8 independent WMMA accumulator chains fill the WMMA->VALU hazard slots.
// block = 256 threads (8 waves), blockIdx.y = channel, 8 pairs/wave.
// ======================================================================
__global__ __launch_bounds__(256) void hopfield_kernel(
    const float* __restrict__ q,     // [64][65536]
    const float* __restrict__ binp,  // [64][64]   (bin_proj[:,:,0])
    const float* __restrict__ wm,    // [64][64][64]
    float* __restrict__ y,           // [64][65536]
    double* __restrict__ stats)      // [2] sum, sumsq
{
  __shared__ float  sBp[64];
  __shared__ double sSum[8], sSq[8];
  const int tid  = threadIdx.x;
  const int lane = tid & 31;
  const int wid  = tid >> 5;
  const int c    = blockIdx.y;

  if (tid < 64) sBp[tid] = binp[c*64 + tid];
  __syncthreads();

  const int lm     = lane & 15;
  const int hiHalf = (lane >> 4) & 1;   // 0: lanes 0-15, 1: lanes 16-31
  const int eoff   = hiHalf ? 8 : 0;

  // ---- A fragments once: W[c] 4 M-tiles x 2 K-chunks, hi/lo bf16 ----
  v16bf Ahi[4][2], Alo[4][2];
  const float* Wc = wm + c*64*64;
  #pragma unroll
  for (int t = 0; t < 4; ++t){
    const float* wrow = Wc + (t*16 + lm)*64;
    #pragma unroll
    for (int kc = 0; kc < 2; ++kc){
      const int eb = kc*32 + eoff;
      #pragma unroll
      for (int j = 0; j < 8; ++j){
        float w0 = wrow[eb + j];
        float w1 = wrow[eb + 16 + j];
        __bf16 h0 = (__bf16)w0;                 // RNE hi part
        __bf16 h1 = (__bf16)w1;
        Ahi[t][kc][j]     = h0;
        Ahi[t][kc][8 + j] = h1;
        Alo[t][kc][j]     = (__bf16)(w0 - (float)h0);   // exact residual
        Alo[t][kc][8 + j] = (__bf16)(w1 - (float)h1);
      }
    }
  }

  // ---- bp values this lane needs (positions match both B layout and the
  //      accumulator D layout):  bpv[t*8+r] = bp[t*16 + eoff + r] ----
  float bpv[32];
  #pragma unroll
  for (int t = 0; t < 4; ++t)
    #pragma unroll
    for (int r = 0; r < 8; ++r)
      bpv[t*8 + r] = sBp[t*16 + eoff + r];

  // packed sign(bp) pattern for the initial state (same index space)
  v8u pos0, pos1;
  #pragma unroll
  for (int r = 0; r < 4; ++r){
    pos0[r]     = pk2sgn(bpv[     2*r], bpv[     2*r + 1]);
    pos0[4 + r] = pk2sgn(bpv[ 8 + 2*r], bpv[ 8 + 2*r + 1]);
    pos1[r]     = pk2sgn(bpv[16 + 2*r], bpv[16 + 2*r + 1]);
    pos1[4 + r] = pk2sgn(bpv[24 + 2*r], bpv[24 + 2*r + 1]);
  }

  double lsum = 0.0, lsq = 0.0;
  const int gwave  = blockIdx.x * 8 + wid;
  const int nwaves = gridDim.x * 8;
  const int npairs = HW / 32;           // 2048 pairs of 16-pixel tiles

  for (int pair = gwave; pair < npairs; pair += nwaves){
    const int   pA = pair*32 + lm;      // tile A pixels
    const int   pB = pA + 16;           // tile B pixels
    const float qA = q[c*HW + pA];
    const float qB = q[c*HW + pB];
    __builtin_prefetch(&q[c*HW + pA + nwaves*32], 0, 1);  // next pair (stream)

    // initial states: flip all bf16 sign bits iff q<0
    const unsigned smA = (unsigned)(((int)__float_as_uint(qA)) >> 31) & 0x80008000u;
    const unsigned smB = (unsigned)(((int)__float_as_uint(qB)) >> 31) & 0x80008000u;
    v8u uA0, uA1, uB0, uB1;
    #pragma unroll
    for (int r = 0; r < 8; ++r){
      uA0[r] = pos0[r] ^ smA;  uA1[r] = pos1[r] ^ smA;
      uB0[r] = pos0[r] ^ smB;  uB1[r] = pos1[r] ^ smB;
    }
    v16bf bA0 = __builtin_bit_cast(v16bf, uA0);
    v16bf bA1 = __builtin_bit_cast(v16bf, uA1);
    v16bf bB0 = __builtin_bit_cast(v16bf, uB0);
    v16bf bB1 = __builtin_bit_cast(v16bf, uB1);

    v8f aA0, aA1, aA2, aA3, aB0, aB1, aB2, aB3;
    const v8f vzero = {};
    #pragma unroll
    for (int it = 0; it < NUM_ITERS; ++it){
      // interleave the two tiles: 8 independent accumulator chains
      aA0 = WMMA_BF16(Ahi[0][0], bA0, vzero);
      aB0 = WMMA_BF16(Ahi[0][0], bB0, vzero);
      aA1 = WMMA_BF16(Ahi[1][0], bA0, vzero);
      aB1 = WMMA_BF16(Ahi[1][0], bB0, vzero);
      aA2 = WMMA_BF16(Ahi[2][0], bA0, vzero);
      aB2 = WMMA_BF16(Ahi[2][0], bB0, vzero);
      aA3 = WMMA_BF16(Ahi[3][0], bA0, vzero);
      aB3 = WMMA_BF16(Ahi[3][0], bB0, vzero);

      aA0 = WMMA_BF16(Alo[0][0], bA0, aA0);
      aB0 = WMMA_BF16(Alo[0][0], bB0, aB0);
      aA1 = WMMA_BF16(Alo[1][0], bA0, aA1);
      aB1 = WMMA_BF16(Alo[1][0], bB0, aB1);
      aA2 = WMMA_BF16(Alo[2][0], bA0, aA2);
      aB2 = WMMA_BF16(Alo[2][0], bB0, aB2);
      aA3 = WMMA_BF16(Alo[3][0], bA0, aA3);
      aB3 = WMMA_BF16(Alo[3][0], bB0, aB3);

      aA0 = WMMA_BF16(Ahi[0][1], bA1, aA0);
      aB0 = WMMA_BF16(Ahi[0][1], bB1, aB0);
      aA1 = WMMA_BF16(Ahi[1][1], bA1, aA1);
      aB1 = WMMA_BF16(Ahi[1][1], bB1, aB1);
      aA2 = WMMA_BF16(Ahi[2][1], bA1, aA2);
      aB2 = WMMA_BF16(Ahi[2][1], bB1, aB2);
      aA3 = WMMA_BF16(Ahi[3][1], bA1, aA3);
      aB3 = WMMA_BF16(Ahi[3][1], bB1, aB3);

      aA0 = WMMA_BF16(Alo[0][1], bA1, aA0);
      aB0 = WMMA_BF16(Alo[0][1], bB1, aB0);
      aA1 = WMMA_BF16(Alo[1][1], bA1, aA1);
      aB1 = WMMA_BF16(Alo[1][1], bB1, aB1);
      aA2 = WMMA_BF16(Alo[2][1], bA1, aA2);
      aB2 = WMMA_BF16(Alo[2][1], bB1, aB2);
      aA3 = WMMA_BF16(Alo[3][1], bA1, aA3);
      aB3 = WMMA_BF16(Alo[3][1], bB1, aB3);

      if (it < NUM_ITERS - 1){
        // repack: D of (a0,a1) -> B chunk 0, (a2,a3) -> B chunk 1
        v8u nA0, nA1, nB0, nB1;
        #pragma unroll
        for (int r = 0; r < 4; ++r){
          nA0[r]     = pk2sgn(aA0[2*r], aA0[2*r + 1]);
          nA0[4 + r] = pk2sgn(aA1[2*r], aA1[2*r + 1]);
          nA1[r]     = pk2sgn(aA2[2*r], aA2[2*r + 1]);
          nA1[4 + r] = pk2sgn(aA3[2*r], aA3[2*r + 1]);
          nB0[r]     = pk2sgn(aB0[2*r], aB0[2*r + 1]);
          nB0[4 + r] = pk2sgn(aB1[2*r], aB1[2*r + 1]);
          nB1[r]     = pk2sgn(aB2[2*r], aB2[2*r + 1]);
          nB1[4 + r] = pk2sgn(aB3[2*r], aB3[2*r + 1]);
        }
        bA0 = __builtin_bit_cast(v16bf, nA0);
        bA1 = __builtin_bit_cast(v16bf, nA1);
        bB0 = __builtin_bit_cast(v16bf, nB0);
        bB1 = __builtin_bit_cast(v16bf, nB1);
      }
    }

    // out[p] = sum_d bp[d]*sign(acc[d,p]) ; sign applied by bit transfer
    float partA = 0.f, partB = 0.f;
    #pragma unroll
    for (int r = 0; r < 8; ++r){
      partA += sgnmul(bpv[     r], aA0[r]);
      partA += sgnmul(bpv[ 8 + r], aA1[r]);
      partA += sgnmul(bpv[16 + r], aA2[r]);
      partA += sgnmul(bpv[24 + r], aA3[r]);
      partB += sgnmul(bpv[     r], aB0[r]);
      partB += sgnmul(bpv[ 8 + r], aB1[r]);
      partB += sgnmul(bpv[16 + r], aB2[r]);
      partB += sgnmul(bpv[24 + r], aB3[r]);
    }
    partA += __shfl_xor(partA, 16, 32);   // combine d-halves
    partB += __shfl_xor(partB, 16, 32);

    const float yA = partA + qA;
    const float yB = partB + qB;
    if (!hiHalf){
      y[c*HW + pA] = yA;
      y[c*HW + pB] = yB;
      lsum += (double)yA + (double)yB;
      lsq  += (double)yA*(double)yA + (double)yB*(double)yB;
    }
  }

  // ---- block reduction of mean/var partials, two double atomics per block ----
  #pragma unroll
  for (int off = 16; off > 0; off >>= 1){
    lsum += __shfl_down(lsum, off, 32);
    lsq  += __shfl_down(lsq,  off, 32);
  }
  if (lane == 0){ sSum[wid] = lsum; sSq[wid] = lsq; }
  __syncthreads();
  if (tid == 0){
    double a = 0.0, b = 0.0;
    #pragma unroll
    for (int i = 0; i < 8; ++i){ a += sSum[i]; b += sSq[i]; }
    atomicAdd(&stats[0], a);
    atomicAdd(&stats[1], b);
  }
}

// ======================================================================
// Kernel 3: global-norm + affine + tanh-GeLU (memory bound).
// ======================================================================
__global__ __launch_bounds__(256) void finalize_kernel(
    const float* __restrict__ y, const double* __restrict__ stats,
    const float* __restrict__ gnw, const float* __restrict__ gnb,
    float* __restrict__ out, int n)
{
  const double N   = (double)n;
  const double m   = stats[0] / N;
  const double var = stats[1] / N - m*m;
  const float mean = (float)m;
  const float rstd = rsqrtf((float)var + 1e-5f);
  for (int i = blockIdx.x*256 + threadIdx.x; i < n; i += gridDim.x*256){
    const int ch = i >> 16;                        // 65536 px per channel
    float v = (y[i] - mean) * rstd * gnw[ch] + gnb[ch];
    float t = 0.7978845608028654f * (v + 0.044715f * v * v * v);
    out[i] = 0.5f * v * (1.f + tanhf(t));
  }
}

// ======================================================================
extern "C" void kernel_launch(void* const* d_in, const int* in_sizes, int n_in,
                              void* d_out, int out_size, void* d_ws, size_t ws_size,
                              hipStream_t stream)
{
  const float* x   = (const float*)d_in[0];   // (64,256,256)
  const float* cw  = (const float*)d_in[1];   // (64,64,3,3)
  const float* cb  = (const float*)d_in[2];   // (64,)
  const float* bp  = (const float*)d_in[3];   // (64,64,1)
  const float* wm  = (const float*)d_in[4];   // (64,64,64)
  const float* gw  = (const float*)d_in[5];   // (64,)
  const float* gb  = (const float*)d_in[6];   // (64,)
  float* out = (float*)d_out;

  char*   ws    = (char*)d_ws;
  float*  q     = (float*)ws;                                   // 16 MB
  float*  y     = (float*)(ws + (size_t)CH*HW*sizeof(float));   // 16 MB
  double* stats = (double*)(ws + 2ull*CH*HW*sizeof(float));     // 16 B

  hipMemsetAsync(stats, 0, 2*sizeof(double), stream);

  conv3x3_kernel <<<dim3(16, 16, 8), 256, 0, stream>>>(x, cw, cb, q);
  hopfield_kernel<<<dim3(32, 64),    256, 0, stream>>>(q, bp, wm, y, stats);
  finalize_kernel<<<dim3(4096),      256, 0, stream>>>(y, stats, gw, gb, out, CH*HW);
}
